// MolGraphIQ_36721970380818
// MI455X (gfx1250) — matrix operand order repaired and checked by reading, not compiled, not run on hardware
//
#include <hip/hip_runtime.h>

// ---------------------------------------------------------------------------
// MolGraphIQ forward (GIN x5 + Set-Transformer readout) for gfx1250 / MI455X.
//
//  * All dense GEMMs -> v_wmma_f32_16x16x32_bf16 (f32 accumulate). Weights
//    pre-transposed+converted once per launch to bf16 [N][Kp]; activations
//    converted to bf16 [M][Kp] before each GEMM. K is PADDED to a multiple of
//    32 and zero-filled, so the GEMM K-loop has NO bounds checks at all:
//    staging is pure global_load_b64 -> ds_store_b64 (compiler cannot
//    re-introduce exec-mask predication). bf16 operands also halve GEMM HBM
//    traffic (~1.5-2 GB total => <100 us @ 23.3 TB/s).
//  * LDS tiles stored in the WMMA per-lane register layout: each lane's
//    fragment is one 32B contiguous LDS read (2x ds_load_b128).
//  * Edge encoder FUSED into the message/scatter kernel (11->300 expansion
//    recomputed per layer, 11x300 W in LDS): avoids a 314 MB edge-feature
//    buffer and ~1.9 GB of traffic.
//  * Attention (<=64 keys, head-dim 75) is tiny -> VALU kernel, K/V + scores
//    staged in LDS.
//
// Input flattening assumption: d_in follows setup_inputs() dict insertion
// order at top level (x, edge_index, edge_attr, batch, params), with 'params'
// flattened as a JAX pytree (dict keys sorted, lists in order).
// ---------------------------------------------------------------------------

#define Bg    512
#define NPGc  64
#define Nn    (Bg * NPGc)     // 32768 nodes
#define Eg    (Nn * 8)        // 262144 edges
#define NDc   30
#define EDc   11
#define Hc    300
#define H2c   600
#define HDc   75
#define HEADSc 4
#define Lc    5
#define KP(k) (((k) + 31) & ~31)      // pad reduction dim to 32

typedef __attribute__((ext_vector_type(16))) __bf16          v16bf;
typedef __attribute__((ext_vector_type(8)))  float           v8f;

__device__ inline unsigned short f2bf(float f) {
    unsigned u = __float_as_uint(f);
    unsigned r = u + 0x7FFFu + ((u >> 16) & 1u);   // round-to-nearest-even
    return (unsigned short)(r >> 16);
}

// ---------------------------------------------------------------------------
// WMMA GEMM on padded bf16 inputs (Kp % 32 == 0, zero-filled pad):
//   C[M,N](f32) = act( A[*,Kp] @ Bt[*,Kp]^T + bias[boff..] (+ resid) )
// Block = 128 threads = 4 waves; tile 64x64, K-step 32.
// LDS tiles pre-permuted to the WMMA per-lane register layout:
//   A (16x32): lane = ((k>>3)&1)*16 + m, elem = (k&7) + ((k>>4)<<3)
//   B (32x16): lane = ((k>>3)&1)*16 + n, elem = (k&7) + ((k>>4)<<3)
// Staging: one k-quad per thread-step: global_load_b64 -> ds_store_b64,
// completely unconditional (rows past M/N are readable slack; they only feed
// outputs the epilogue never stores).
// ---------------------------------------------------------------------------
__global__ __launch_bounds__(128)
void gemm_wmma(const unsigned short* __restrict__ A,    // [>=ceil64(M)][Kp]
               const unsigned short* __restrict__ Bt,   // [>=ceil64(N)][Kp]
               const float* __restrict__ bias, int boff,
               const float* __restrict__ resid, int ldr,
               float* __restrict__ C, int ldc,
               int M, int N, int Kp, int relu)
{
    __shared__ unsigned short Af[4][32][16];   // [m-subtile][lane][elem] 4KB
    __shared__ unsigned short Bf[4][32][16];   // [n-subtile][lane][elem] 4KB

    const int tid  = threadIdx.x;
    const int lane = tid & 31;
    const int wave = tid >> 5;
    const int m0 = blockIdx.y * 64;
    const int n0 = blockIdx.x * 64;

    v8f acc[4];
    #pragma unroll
    for (int t = 0; t < 4; ++t)
        #pragma unroll
        for (int r = 0; r < 8; ++r) acc[t][r] = 0.0f;

    const int nk = Kp >> 5;
    for (int kt = 0; kt < nk; ++kt) {
        const int k0 = kt * 32;

        // ---- stage A tile 64x32: 4 unconditional b64 quads per thread ----
        #pragma unroll
        for (int i = 0; i < 4; ++i) {
            int q  = tid + i * 128;
            int r  = q >> 3;            // 0..63 (8 threads sweep one row)
            int kq = (q & 7) << 2;      // 0,4,...,28
            uint2 val = *(const uint2*)(A + (size_t)(m0 + r) * Kp + k0 + kq);
            int half = (kq >> 3) & 1;
            int elem = (kq & 7) + ((kq >> 4) << 3);
            *(uint2*)&Af[r >> 4][half * 16 + (r & 15)][elem] = val;
        }
        // ---- stage B tile 64n x 32k ----
        #pragma unroll
        for (int i = 0; i < 4; ++i) {
            int q  = tid + i * 128;
            int n  = q >> 3;
            int kq = (q & 7) << 2;
            uint2 val = *(const uint2*)(Bt + (size_t)(n0 + n) * Kp + k0 + kq);
            int half = (kq >> 3) & 1;
            int elem = (kq & 7) + ((kq >> 4) << 3);
            *(uint2*)&Bf[n >> 4][half * 16 + (n & 15)][elem] = val;
        }
        __syncthreads();

        // ---- fragment loads: lane-contiguous 32B (2x ds_load_b128 each) ----
        v16bf a  = *(const v16bf*)&Af[wave][lane][0];
        v16bf b0 = *(const v16bf*)&Bf[0][lane][0];
        v16bf b1 = *(const v16bf*)&Bf[1][lane][0];
        v16bf b2 = *(const v16bf*)&Bf[2][lane][0];
        v16bf b3 = *(const v16bf*)&Bf[3][lane][0];

        // 4 independent accumulators -> back-to-back WMMAs (no D->A/B hazard)
        acc[0] = __builtin_amdgcn_wmma_f32_16x16x32_bf16(false, a, false, b0,
                                                         (short)0, acc[0], false, false);
        acc[1] = __builtin_amdgcn_wmma_f32_16x16x32_bf16(false, a, false, b1,
                                                         (short)0, acc[1], false, false);
        acc[2] = __builtin_amdgcn_wmma_f32_16x16x32_bf16(false, a, false, b2,
                                                         (short)0, acc[2], false, false);
        acc[3] = __builtin_amdgcn_wmma_f32_16x16x32_bf16(false, a, false, b3,
                                                         (short)0, acc[3], false, false);
        __syncthreads();
    }

    // ---- epilogue: C/D layout -> row (lane>=16)*8 + r, col lane%16 ----
    const int half = lane >> 4;
    const int ln16 = lane & 15;
    if ((m0 + 64 <= M) && (n0 + 64 <= N)) {          // uniform interior path
        #pragma unroll
        for (int t = 0; t < 4; ++t) {
            int gn = n0 + t * 16 + ln16;
            float bv = bias ? bias[boff + gn] : 0.0f;
            #pragma unroll
            for (int r = 0; r < 8; ++r) {
                int gm = m0 + wave * 16 + half * 8 + r;
                float x = acc[t][r] + bv;
                if (resid) x += resid[(size_t)gm * ldr + gn];
                if (relu)  x = fmaxf(x, 0.0f);
                C[(size_t)gm * ldc + gn] = x;
            }
        }
    } else {                                          // ragged blocks only
        #pragma unroll
        for (int t = 0; t < 4; ++t) {
            int gn = n0 + t * 16 + ln16;
            if (gn >= N) continue;
            float bv = bias ? bias[boff + gn] : 0.0f;
            #pragma unroll
            for (int r = 0; r < 8; ++r) {
                int gm = m0 + wave * 16 + half * 8 + r;
                if (gm >= M) continue;
                float x = acc[t][r] + bv;
                if (resid) x += resid[(size_t)gm * ldr + gn];
                if (relu)  x = fmaxf(x, 0.0f);
                C[(size_t)gm * ldc + gn] = x;
            }
        }
    }
}

// f32 [M,K] dense -> bf16 [M,Kp] with zero-filled pad columns
__global__ void cvt_pad(const float* __restrict__ in, unsigned* __restrict__ out,
                        int M, int K, int Kp)
{
    const int kpo = Kp >> 1;
    size_t total = (size_t)M * kpo;
    for (size_t idx = blockIdx.x * (size_t)blockDim.x + threadIdx.x; idx < total;
         idx += (size_t)gridDim.x * blockDim.x) {
        int c   = (int)(idx % kpo);
        int row = (int)(idx / kpo);
        int k = c * 2;
        unsigned v = 0u;
        if (k < K) {        // K is even in all uses -> pair fully in or out
            const float* p = in + (size_t)row * K + k;
            v = (unsigned)f2bf(p[0]) | ((unsigned)f2bf(p[1]) << 16);
        }
        out[idx] = v;
    }
}

// Weight transpose+convert: Wt[n*Kp + k] = bf16( W[k*ldw + woff + n] ), 0-pad k
__global__ __launch_bounds__(256)
void wtrans_cvt(const float* __restrict__ W, int ldw, int woff,
                unsigned short* __restrict__ Wt, int K, int N, int Kp)
{
    __shared__ float t[32][33];
    int kb = blockIdx.x * 32;
    int nb = blockIdx.y * 32;
    for (int i = threadIdx.y; i < 32; i += blockDim.y) {
        int k = kb + i, n = nb + threadIdx.x;
        t[i][threadIdx.x] = (k < K && n < N) ? W[(size_t)k * ldw + woff + n] : 0.0f;
    }
    __syncthreads();
    for (int i = threadIdx.y; i < 32; i += blockDim.y) {
        int n = nb + i, k = kb + threadIdx.x;
        if (n < N && k < Kp) Wt[(size_t)n * Kp + k] = f2bf(t[threadIdx.x][i]);
    }
}

// ---------------------------------------------------------------------------
// Fused edge-encode + message + scatter:
//   agg[dst] += relu( h[src] + edge_attr @ We + be )
// ---------------------------------------------------------------------------
__global__ __launch_bounds__(256)
void msg_scatter(const float* __restrict__ h,
                 const float* __restrict__ eattr,
                 const int* __restrict__ src,
                 const int* __restrict__ dst,
                 const float* __restrict__ We,   // 11 x 300
                 const float* __restrict__ be,   // 300
                 float* __restrict__ agg, int E)
{
    __shared__ float WeS[EDc][Hc];
    __shared__ float beS[Hc];
    for (int i = threadIdx.x; i < EDc * Hc; i += blockDim.x)
        WeS[i / Hc][i % Hc] = We[i];
    for (int i = threadIdx.x; i < Hc; i += blockDim.x) beS[i] = be[i];
    __syncthreads();

    const int lane = threadIdx.x & 31;
    const int warp = threadIdx.x >> 5;
    const int wpb  = blockDim.x >> 5;
    for (int e = blockIdx.x * wpb + warp; e < E; e += gridDim.x * wpb) {
        int s = src[e], d = dst[e];
        float at[EDc];
        #pragma unroll
        for (int t = 0; t < EDc; ++t) at[t] = eattr[(size_t)e * EDc + t];
        const float* hs = h + (size_t)s * Hc;
        float* ag = agg + (size_t)d * Hc;
        __builtin_prefetch(hs, 0, 1);           // global_prefetch_b8
        for (int j = lane; j < Hc; j += 32) {
            float ea = beS[j];
            #pragma unroll
            for (int t = 0; t < EDc; ++t) ea += at[t] * WeS[t][j];
            float m = fmaxf(hs[j] + ea, 0.0f);
            atomicAdd(&ag[j], m);
        }
    }
}

// z = (1+eps)*h + agg   (may alias z==agg)
__global__ void zcomb(const float* __restrict__ h, const float* __restrict__ agg,
                      const float* __restrict__ epsp, float* __restrict__ z,
                      size_t n)
{
    float c = 1.0f + *epsp;
    for (size_t i = blockIdx.x * (size_t)blockDim.x + threadIdx.x; i < n;
         i += (size_t)gridDim.x * blockDim.x)
        z[i] = c * h[i] + agg[i];
}

__global__ __launch_bounds__(256)
void bn_stats(const float* __restrict__ z, float* __restrict__ mu,
              float* __restrict__ rstd, int M)
{
    int j = blockIdx.x;
    float s = 0.0f, s2 = 0.0f;
    for (int i = threadIdx.x; i < M; i += blockDim.x) {
        float v = z[(size_t)i * Hc + j];
        s += v; s2 += v * v;
    }
    __shared__ float sh[256], sh2[256];
    sh[threadIdx.x] = s; sh2[threadIdx.x] = s2;
    __syncthreads();
    for (int o = 128; o > 0; o >>= 1) {
        if (threadIdx.x < o) {
            sh[threadIdx.x]  += sh[threadIdx.x + o];
            sh2[threadIdx.x] += sh2[threadIdx.x + o];
        }
        __syncthreads();
    }
    if (threadIdx.x == 0) {
        float m = sh[0] / M;
        float var = sh2[0] / M - m * m;
        mu[j] = m;
        rstd[j] = rsqrtf(var + 1e-5f);
    }
}

__global__ void bn_apply(const float* __restrict__ z, const float* __restrict__ mu,
                         const float* __restrict__ rstd,
                         const float* __restrict__ gamma,
                         const float* __restrict__ beta,
                         float* __restrict__ h, int M)
{
    size_t n = (size_t)M * Hc;
    for (size_t i = blockIdx.x * (size_t)blockDim.x + threadIdx.x; i < n;
         i += (size_t)gridDim.x * blockDim.x) {
        int j = (int)(i % Hc);
        float v = gamma[j] * (z[i] - mu[j]) * rstd[j] + beta[j];
        h[i] = fmaxf(v, 0.0f);
    }
}

// ---------------------------------------------------------------------------
// Multi-head attention: block = (graph, head), 64 threads = query rows.
// ---------------------------------------------------------------------------
__global__ __launch_bounds__(64)
void attention(const float* __restrict__ q, const float* __restrict__ k,
               const float* __restrict__ v, float* __restrict__ o,
               int Lq, int Lk)
{
    const int b  = blockIdx.x;
    const int hh = blockIdx.y;
    __shared__ float Ks[64][HDc + 1];
    __shared__ float Vs[64][HDc + 1];
    __shared__ float Ss[64][65];

    for (int i = threadIdx.x; i < Lk * HDc; i += blockDim.x) {
        int r = i / HDc, c = i % HDc;
        size_t base = ((size_t)b * Lk + r) * Hc + hh * HDc + c;
        Ks[r][c] = k[base];
        Vs[r][c] = v[base];
    }
    __syncthreads();

    int i = threadIdx.x;
    if (i >= Lq) return;

    float qr[HDc];
    #pragma unroll
    for (int c = 0; c < HDc; ++c)
        qr[c] = q[((size_t)b * Lq + i) * Hc + hh * HDc + c];

    const float scale = 0.115470053837925f;   // 1/sqrt(75)
    float mx = -3.4e38f;
    for (int kk = 0; kk < Lk; ++kk) {
        float s = 0.0f;
        #pragma unroll
        for (int c = 0; c < HDc; ++c) s += qr[c] * Ks[kk][c];
        s *= scale;
        Ss[i][kk] = s;
        mx = fmaxf(mx, s);
    }
    float ssum = 0.0f;
    for (int kk = 0; kk < Lk; ++kk) {
        float e = __expf(Ss[i][kk] - mx);
        Ss[i][kk] = e;
        ssum += e;
    }
    float inv = 1.0f / ssum;
    #pragma unroll
    for (int c = 0; c < HDc; ++c) {
        float acc = 0.0f;
        for (int kk = 0; kk < Lk; ++kk) acc += Ss[i][kk] * Vs[kk][c];
        o[((size_t)b * Lq + i) * Hc + hh * HDc + c] = acc * inv;
    }
}

__global__ void bcast_row(const float* __restrict__ row, float* __restrict__ out,
                          int rows, int cols)
{
    int n = rows * cols;
    for (int i = blockIdx.x * blockDim.x + threadIdx.x; i < n;
         i += gridDim.x * blockDim.x)
        out[i] = row[i % cols];
}

// ---------------------------------------------------------------------------
// Host-side orchestration
// ---------------------------------------------------------------------------
static void gemm(hipStream_t st, const unsigned short* A,
                 const unsigned short* Bt,
                 const float* bias, int boff,
                 const float* resid, int ldr,
                 float* C, int ldc, int M, int N, int Kp, int relu)
{
    dim3 g((N + 63) / 64, (M + 63) / 64);
    hipLaunchKernelGGL(gemm_wmma, g, dim3(128), 0, st,
                       A, Bt, bias, boff, resid, ldr, C, ldc, M, N, Kp, relu);
}

static void cvt(hipStream_t st, const float* in, unsigned short* out, int M, int K)
{
    hipLaunchKernelGGL(cvt_pad, dim3(2048), dim3(256), 0, st,
                       in, (unsigned*)out, M, K, KP(K));
}

struct MabW {
    const unsigned short *wq, *wk, *wv, *wo, *wl;
    const float *bi, *bo, *lb;
};

// One MAB: out = relu( lin( MHA(Q,K) + Q ) ).  Qbf/Kbf bf16 [*,320];
// Qf is the f32 residual. abf is scratch (overwritten after attention).
static void mab(hipStream_t st, const float* Qf, const unsigned short* Qbf, int Lq,
                const unsigned short* Kbf, int Lk, const MabW& Wm,
                float* bq, float* bk, float* bv, float* bob, float* btmp,
                unsigned short* abf, float* outp)
{
    const int Mq = Bg * Lq, Mk = Bg * Lk;
    const int Kp = KP(Hc);
    gemm(st, Qbf, Wm.wq, Wm.bi, 0,      nullptr, 0, bq, Hc, Mq, Hc, Kp, 0);
    gemm(st, Kbf, Wm.wk, Wm.bi, Hc,     nullptr, 0, bk, Hc, Mk, Hc, Kp, 0);
    gemm(st, Kbf, Wm.wv, Wm.bi, 2 * Hc, nullptr, 0, bv, Hc, Mk, Hc, Kp, 0);
    hipLaunchKernelGGL(attention, dim3(Bg, HEADSc), dim3(64), 0, st,
                       bq, bk, bv, bob, Lq, Lk);
    cvt(st, bob, abf, Mq, Hc);
    gemm(st, abf, Wm.wo, Wm.bo, 0, Qf, Hc, btmp, Hc, Mq, Hc, Kp, 0);
    cvt(st, btmp, abf, Mq, Hc);
    gemm(st, abf, Wm.wl, Wm.lb, 0, nullptr, 0, outp, Hc, Mq, Hc, Kp, 1);
}

extern "C" void kernel_launch(void* const* d_in, const int* in_sizes, int n_in,
                              void* d_out, int out_size, void* d_ws, size_t ws_size,
                              hipStream_t stream)
{
    (void)in_sizes; (void)n_in; (void)out_size; (void)ws_size;

    // --- input leaves (see flattening note at top of file) ---
    const float* x      = (const float*)d_in[0];                 // 32768 x 30
    const int*   ei     = (const int*)  d_in[1];                 // 2 x 262144
    const float* eattr  = (const float*)d_in[2];                 // 262144 x 11
    const float* auxW0 = (const float*)d_in[4],  *auxB0 = (const float*)d_in[5];
    const float* auxW1 = (const float*)d_in[6],  *auxB1 = (const float*)d_in[7];
    const float* edgeW = (const float*)d_in[43], *edgeB = (const float*)d_in[44];
    const float* nodeW = (const float*)d_in[45], *nodeB = (const float*)d_in[46];
    const float* primW0 = (const float*)d_in[47], *primB0 = (const float*)d_in[48];
    const float* primW1 = (const float*)d_in[49], *primB1 = (const float*)d_in[50];
    const float* seedp  = (const float*)d_in[77];                // 300

    const int* srcI = ei;
    const int* dstI = ei + Eg;

    // --- workspace layout (f32 part; all sizes even) ---
    float* w = (float*)d_ws;
    size_t off = 0;
    const size_t NH = (size_t)Nn * Hc;
    float* bufh   = w + off; off += NH;
    float* buf600 = w + off; off += (size_t)Nn * H2c;
    float* bufagg = w + off; off += NH;
    float* bufq   = w + off; off += NH;
    float* bufk   = w + off; off += NH;
    float* bufv   = w + off; off += NH;
    float* bufo   = w + off; off += NH;
    float* bufseeds = w + off; off += (size_t)Bg * Hc;
    float* bufhid   = w + off; off += (size_t)Bg * 150;
    float* srow  = w + off; off += 512;
    float* muB   = w + off; off += 512;
    float* rstdB = w + off; off += 512;

    unsigned short* wbf = (unsigned short*)(w + off);     // bf16 weight arena
    size_t wo = 0;
    auto wt = [&](const float* Wp, int ldw, int woff2, int K, int N)
              -> const unsigned short* {
        int Kp = KP(K);
        unsigned short* dstp = wbf + wo;
        wo += (size_t)Kp * N;
        dim3 g((Kp + 31) / 32, (N + 31) / 32);
        hipLaunchKernelGGL(wtrans_cvt, g, dim3(32, 8), 0, stream,
                           Wp, ldw, woff2, dstp, K, N, Kp);
        return dstp;
    };

    // ---- pre-transpose+convert all weights (bf16, [N][Kp]) ----
    const unsigned short* nodeWt = wt(nodeW, Hc, 0, NDc, Hc);
    const unsigned short *l1Wt[Lc], *l2Wt[Lc];
    for (int l = 0; l < Lc; ++l) {
        l1Wt[l] = wt((const float*)d_in[19 + 5 * l], H2c, 0, Hc, H2c);
        l2Wt[l] = wt((const float*)d_in[21 + 5 * l], Hc, 0, H2c, Hc);
    }
    auto mkmab = [&](int iWi, int iWo, int ibi, int ibo, int ilW, int ilb) {
        MabW m;
        const float* Wi = (const float*)d_in[iWi];
        m.wq = wt(Wi, 3 * Hc, 0,      Hc, Hc);
        m.wk = wt(Wi, 3 * Hc, Hc,     Hc, Hc);
        m.wv = wt(Wi, 3 * Hc, 2 * Hc, Hc, Hc);
        m.wo = wt((const float*)d_in[iWo], Hc, 0, Hc, Hc);
        m.wl = wt((const float*)d_in[ilW], Hc, 0, Hc, Hc);
        m.bi = (const float*)d_in[ibi];
        m.bo = (const float*)d_in[ibo];
        m.lb = (const float*)d_in[ilb];
        return m;
    };
    MabW encM[2] = { mkmab(57, 58, 59, 60, 61, 62),
                     mkmab(63, 64, 65, 66, 67, 68) };
    MabW pmaM = mkmab(71, 72, 73, 74, 75, 76);
    MabW decM = mkmab(51, 52, 53, 54, 55, 56);
    const unsigned short* pmaLint = wt((const float*)d_in[69], Hc, 0, Hc, Hc);
    const float* pmaLinB = (const float*)d_in[70];
    const unsigned short* prim0t = wt(primW0, 150, 0, Hc, 150);
    const unsigned short* prim1t = wt(primW1, 1,   0, 150, 1);
    const unsigned short* aux0t  = wt(auxW0,  150, 0, Hc, 150);
    const unsigned short* aux1t  = wt(auxW1,  82,  0, 150, 82);
    wo += (size_t)64 * KP(150);                           // N-overread slack

    unsigned short* abf = wbf + ((wo + 3) & ~(size_t)3);  // bf16 activations
    unsigned short* sbf = abf + (size_t)Nn * KP(H2c);     // small bf16 scratch
    // (sbf sized Bg*KP(Hc) + 64*KP(Hc) slack; all within harness workspace)

    float* outPrim = (float*)d_out;
    float* outAux  = outPrim + Bg;
    float* outG    = outAux + (size_t)Bg * 82;

    const dim3 EB(2048), ET(256);
    const int KpH = KP(Hc);                               // 320

    // ---- node encoder ----
    cvt(stream, x, abf, Nn, NDc);
    gemm(stream, abf, nodeWt, nodeB, 0, nullptr, 0, bufh, Hc, Nn, Hc, KP(NDc), 0);

    // ---- 5 GIN conv layers ----
    for (int l = 0; l < Lc; ++l) {
        const float* epsP = (const float*)d_in[18 + 5 * l];
        const float* l1b  = (const float*)d_in[20 + 5 * l];
        const float* l2b  = (const float*)d_in[22 + 5 * l];
        const float* beta  = (const float*)d_in[8 + 2 * l];
        const float* gamma = (const float*)d_in[9 + 2 * l];

        hipMemsetAsync(bufagg, 0, NH * sizeof(float), stream);
        hipLaunchKernelGGL(msg_scatter, EB, ET, 0, stream,
                           bufh, eattr, srcI, dstI, edgeW, edgeB, bufagg, Eg);
        hipLaunchKernelGGL(zcomb, EB, ET, 0, stream, bufh, bufagg, epsP, bufagg, NH);
        cvt(stream, bufagg, abf, Nn, Hc);
        gemm(stream, abf, l1Wt[l], l1b, 0, nullptr, 0,
             buf600, H2c, Nn, H2c, KpH, 1);
        cvt(stream, buf600, abf, Nn, H2c);
        gemm(stream, abf, l2Wt[l], l2b, 0, nullptr, 0,
             bufq, Hc, Nn, Hc, KP(H2c), 0);
        hipLaunchKernelGGL(bn_stats, dim3(Hc), dim3(256), 0, stream,
                           bufq, muB, rstdB, Nn);
        hipLaunchKernelGGL(bn_apply, EB, ET, 0, stream,
                           bufq, muB, rstdB, gamma, beta, bufh, Nn);
    }

    // ---- readout: 2x SAB on hp = bufh (512,64,300) ----
    for (int e = 0; e < 2; ++e) {
        cvt(stream, bufh, abf, Nn, Hc);
        mab(stream, bufh, abf, NPGc, abf, NPGc, encM[e],
            bufq, bufk, bufv, bufo, buf600, abf, bufh);
    }

    // ---- PMA ----
    cvt(stream, seedp, sbf, 1, Hc);
    gemm(stream, sbf, pmaLint, pmaLinB, 0, nullptr, 0, srow, Hc, 1, Hc, KpH, 1);
    hipLaunchKernelGGL(bcast_row, dim3(512), ET, 0, stream, srow, bufseeds, Bg, Hc);
    cvt(stream, bufseeds, sbf, Bg, Hc);
    cvt(stream, bufh, abf, Nn, Hc);
    mab(stream, bufseeds, sbf, 1, abf, NPGc, pmaM,
        bufq, bufk, bufv, bufo, buf600, abf, bufh);     // -> bufh[0:512,:]

    // ---- dec SAB on (512,1,300) -> g region of d_out ----
    cvt(stream, bufh, abf, Bg, Hc);
    mab(stream, bufh, abf, 1, abf, 1, decM,
        bufq, bufk, bufv, bufo, buf600, abf, outG);

    // ---- heads ----
    cvt(stream, outG, abf, Bg, Hc);
    gemm(stream, abf, prim0t, primB0, 0, nullptr, 0, bufhid, 150, Bg, 150, KpH, 1);
    cvt(stream, bufhid, sbf, Bg, 150);
    gemm(stream, sbf, prim1t, primB1, 0, nullptr, 0, outPrim, 1, Bg, 1, KP(150), 0);
    gemm(stream, abf, aux0t, auxB0, 0, nullptr, 0, bufhid, 150, Bg, 150, KpH, 1);
    cvt(stream, bufhid, sbf, Bg, 150);
    gemm(stream, sbf, aux1t, auxB1, 0, nullptr, 0, outAux, 82, Bg, 82, KP(150), 0);
}
